// NewBackboneModel_82849919139915
// MI455X (gfx1250) — matrix operand order repaired and checked by reading, not compiled, not run on hardware
//
#include <hip/hip_runtime.h>
#include <hip/hip_bf16.h>
#include <math.h>

typedef __bf16 bf16_t;
typedef __attribute__((ext_vector_type(16))) __bf16 v16bf;
typedef __attribute__((ext_vector_type(8)))  __bf16 v8bf;
typedef __attribute__((ext_vector_type(8)))  float  v8f;

#define BN_EPS 1e-5f
#define B_   4
#define N_   2048
#define P_   1024
#define Q_   16
#define L_   64
#define V_   64
#define OUT_ 40
#define M_   (Q_*P_)          // 16384 primary-caps channels
#define SAMPLES_ 8192.0f      // B_*N_

// ---------- helpers ----------
__device__ __forceinline__ void atomicMaxF(float* addr, float val){
  if (val >= 0.f) atomicMax((int*)addr, __float_as_int(val));
  else            atomicMin((unsigned int*)addr, __float_as_uint(val));
}
__device__ __forceinline__ void atomicMinF(float* addr, float val){
  if (val >= 0.f) atomicMin((int*)addr, __float_as_int(val));
  else            atomicMax((unsigned int*)addr, __float_as_uint(val));
}
__device__ __forceinline__ unsigned short f2bf(float f){
  unsigned u = __float_as_uint(f);
  u += 0x7FFFu + ((u >> 16) & 1u);       // round-to-nearest-even
  return (unsigned short)(u >> 16);
}

// ---------- init ----------
__global__ void fill_u32_kernel(unsigned* p, unsigned v, int n){
  int i = blockIdx.x*blockDim.x + threadIdx.x;
  if (i < n) p[i] = v;
}

// ---------- stage 1: conv1 (3->64) with fused BN-stat reduction ----------
__global__ __launch_bounds__(256) void conv1_kernel(
    const float* __restrict__ x, const float* __restrict__ w1, const float* __restrict__ b1,
    float* __restrict__ h1pre, float* sum1, float* sq1){
  int n = blockIdx.x*256 + threadIdx.x;
  int c = blockIdx.y, b = blockIdx.z;
  const float* xb = x + (size_t)b*3*N_ + n;
  float v = fmaf(w1[c*3+0], xb[0],
           fmaf(w1[c*3+1], xb[N_],
           fmaf(w1[c*3+2], xb[2*N_], b1[c])));
  h1pre[((size_t)b*64 + c)*N_ + n] = v;
  __shared__ float rs[256], rq[256];
  rs[threadIdx.x] = v; rq[threadIdx.x] = v*v; __syncthreads();
  for (int s = 128; s > 0; s >>= 1){
    if (threadIdx.x < s){ rs[threadIdx.x] += rs[threadIdx.x+s]; rq[threadIdx.x] += rq[threadIdx.x+s]; }
    __syncthreads();
  }
  if (threadIdx.x == 0){ atomicAdd(&sum1[c], rs[0]); atomicAdd(&sq1[c], rq[0]); }
}

// ---------- BN finalize: a = g*rsqrt(var+eps), c = be - mean*a ----------
__global__ void bn_finalize_kernel(const float* sum, const float* sq, const float* g,
                                   const float* be, float* a, float* c, int nch){
  int i = blockIdx.x*blockDim.x + threadIdx.x;
  if (i >= nch) return;
  float mean = sum[i] * (1.f/SAMPLES_);
  float var  = sq[i]  * (1.f/SAMPLES_) - mean*mean;
  float s = g[i] * rsqrtf(var + BN_EPS);
  a[i] = s; c[i] = be[i] - mean*s;
}

__global__ void apply1_kernel(const float* __restrict__ h1pre, const float* __restrict__ a,
                              const float* __restrict__ c, float* __restrict__ h1){
  int i = blockIdx.x*256 + threadIdx.x;
  int ch = (i >> 11) & 63;            // (i / N_) % 64
  h1[i] = fmaxf(fmaf(a[ch], h1pre[i], c[ch]), 0.f);
}

// ---------- stage 1b: conv2 (64->128) with fused BN stats ----------
__global__ __launch_bounds__(256) void conv2_kernel(
    const float* __restrict__ h1, const float* __restrict__ w2, const float* __restrict__ b2,
    float* __restrict__ h2pre, float* sum2, float* sq2){
  int n = blockIdx.x*256 + threadIdx.x;
  int o = blockIdx.y, b = blockIdx.z;
  const float* hb = h1 + (size_t)b*64*N_ + n;
  const float* w  = w2 + o*64;         // uniform per block -> scalar loads
  float acc = b2[o];
  #pragma unroll 8
  for (int c = 0; c < 64; ++c) acc = fmaf(w[c], hb[(size_t)c*N_], acc);
  h2pre[((size_t)b*128 + o)*N_ + n] = acc;
  __shared__ float rs[256], rq[256];
  rs[threadIdx.x] = acc; rq[threadIdx.x] = acc*acc; __syncthreads();
  for (int s = 128; s > 0; s >>= 1){
    if (threadIdx.x < s){ rs[threadIdx.x] += rs[threadIdx.x+s]; rq[threadIdx.x] += rq[threadIdx.x+s]; }
    __syncthreads();
  }
  if (threadIdx.x == 0){ atomicAdd(&sum2[o], rs[0]); atomicAdd(&sq2[o], rq[0]); }
}

// ---------- BN+ReLU, transpose to [b][n][c], convert to bf16 ----------
__global__ void apply2t_kernel(const float* __restrict__ h2pre, const float* __restrict__ a2,
                               const float* __restrict__ c2, unsigned short* __restrict__ hTbf){
  int b = blockIdx.x / N_, n = blockIdx.x % N_, o = threadIdx.x;   // block = 128 threads
  float v = fmaxf(fmaf(a2[o], h2pre[((size_t)b*128 + o)*N_ + n], c2[o]), 0.f);
  hTbf[((size_t)(b*N_ + n))*128 + o] = f2bf(v);
}

__global__ void f32_to_bf16_kernel(const float* __restrict__ in, unsigned short* __restrict__ out, int n){
  int i = blockIdx.x*blockDim.x + threadIdx.x;
  if (i < n) out[i] = f2bf(in[i]);
}

// ---------- THE big GEMM: u[m=q*P+p, j=b*N+n] = w3[m,:128] . hT[j,:128] ----------
// bf16 WMMA 16x16x32, f32 accumulate. Never materializes u: epilogue reduces
// per-row sum/sumsq (BN over B,N) and per-(b,row) max/min (maxpool over N).
// Two column tiles in flight per iteration -> two independent WMMA chains
// (matrix-pipe ILP) and twice the loads outstanding per wait.
__global__ __launch_bounds__(256) void caps_gemm_kernel(
    const unsigned short* __restrict__ w3bf, const unsigned short* __restrict__ hTbf,
    float* __restrict__ sum3, float* __restrict__ sq3,
    float* __restrict__ mx3,  float* __restrict__ mn3){
  int wave = (blockIdx.x*256 + threadIdx.x) >> 5;   // 0..32767
  int lane = threadIdx.x & 31;
  int mtile = wave >> 5;                            // 0..1023
  int job   = wave & 31;
  int b     = job >> 3;                             // batch 0..3
  int grp   = job & 7;                              // column group within batch
  int m0  = mtile * 16;
  int row = lane & 15;
  int sel = lane >> 4;

  // preload A fragments for all 4 K-steps (contiguous K order, matching B:
  // lanes 0-15: K 0-15 ; lanes 16-31: K 16-31)
  const bf16_t* W = (const bf16_t*)w3bf;
  v16bf afr[4];
  #pragma unroll
  for (int ks = 0; ks < 4; ++ks){
    const bf16_t* ap = W + (size_t)(m0 + row)*128 + ks*32 + sel*16;
    v8bf a0 = *(const v8bf*)(ap);
    v8bf a1 = *(const v8bf*)(ap + 8);
    #pragma unroll
    for (int i = 0; i < 8; ++i){ afr[ks][i] = a0[i]; afr[ks][8+i] = a1[i]; }
  }

  float ssum[8], ssq[8], smx[8], smn[8];
  #pragma unroll
  for (int r = 0; r < 8; ++r){ ssum[r]=0.f; ssq[r]=0.f; smx[r]=-__builtin_inff(); smn[r]=__builtin_inff(); }

  const bf16_t* H = (const bf16_t*)hTbf;
  for (int jt = 0; jt < 16; jt += 2){
    int j0 = (b*128 + grp*16 + jt) * 16;            // global column base
    const bf16_t* bp0 = H + (size_t)(j0 + (lane & 15))*128 + sel*16;
    const bf16_t* bp1 = bp0 + (size_t)16*128;       // next 16-column tile
    v8f acc0 = {};
    v8f acc1 = {};
    #pragma unroll
    for (int ks = 0; ks < 4; ++ks){
      v8bf b00 = *(const v8bf*)(bp0 + ks*32);
      v8bf b01 = *(const v8bf*)(bp0 + ks*32 + 8);
      v8bf b10 = *(const v8bf*)(bp1 + ks*32);
      v8bf b11 = *(const v8bf*)(bp1 + ks*32 + 8);
      v16bf f0, f1;
      #pragma unroll
      for (int i = 0; i < 8; ++i){
        f0[i] = b00[i]; f0[8+i] = b01[i];
        f1[i] = b10[i]; f1[8+i] = b11[i];
      }
      acc0 = __builtin_amdgcn_wmma_f32_16x16x32_bf16(false, afr[ks], false, f0,
                                                     (short)0, acc0, false, false);
      acc1 = __builtin_amdgcn_wmma_f32_16x16x32_bf16(false, afr[ks], false, f1,
                                                     (short)0, acc1, false, false);
    }
    #pragma unroll
    for (int r = 0; r < 8; ++r){
      float v0 = acc0[r], v1 = acc1[r];
      ssum[r] += v0 + v1;
      ssq[r]  += v0*v0 + v1*v1;
      smx[r] = fmaxf(smx[r], fmaxf(v0, v1));
      smn[r] = fminf(smn[r], fminf(v0, v1));
    }
  }

  // reduce across the 16 lanes of each half (rows r and r+8 live in halves)
  #pragma unroll
  for (int r = 0; r < 8; ++r){
    float s = ssum[r], q = ssq[r], x = smx[r], n2 = smn[r];
    #pragma unroll
    for (int m = 1; m < 16; m <<= 1){
      s  += __shfl_xor(s,  m, 32);
      q  += __shfl_xor(q,  m, 32);
      x   = fmaxf(x,  __shfl_xor(x,  m, 32));
      n2  = fminf(n2, __shfl_xor(n2, m, 32));
    }
    if ((lane & 15) == 0){
      int mrow = m0 + r + sel*8;
      atomicAdd(&sum3[mrow], s);
      atomicAdd(&sq3[mrow],  q);
      atomicMaxF(&mx3[b*M_ + mrow], x);
      atomicMinF(&mn3[b*M_ + mrow], n2);
    }
  }
}

// ---------- BN finalize for caps + max-through-affine + squash over Q ----------
__global__ void caps_squash_kernel(const float* __restrict__ sum3, const float* __restrict__ sq3,
                                   const float* __restrict__ g3, const float* __restrict__ be3,
                                   const float* __restrict__ mx3, const float* __restrict__ mn3,
                                   float* __restrict__ u){
  int i = blockIdx.x*blockDim.x + threadIdx.x;
  if (i >= B_*P_) return;
  int b = i / P_, p = i % P_;
  float vals[Q_]; float sn = 0.f;
  #pragma unroll
  for (int q = 0; q < Q_; ++q){
    int m = q*P_ + p;
    float mean = sum3[m] * (1.f/SAMPLES_);
    float var  = sq3[m]  * (1.f/SAMPLES_) - mean*mean;
    float a = g3[m] * rsqrtf(var + BN_EPS);
    float c = be3[m] - mean*a;
    float v = (a >= 0.f) ? fmaf(a, mx3[b*M_ + m], c) : fmaf(a, mn3[b*M_ + m], c);
    vals[q] = v; sn += v*v;
  }
  float sc = sn / ((1.f + sn) * sqrtf(sn));
  #pragma unroll
  for (int q = 0; q < Q_; ++q) u[((size_t)b*P_ + p)*Q_ + q] = sc*vals[q];
}

// ---------- u_hat[b,l,p,v] = sum_q Wr[l,p,v,q]*u[b,p,q] (streams 268MB Wr once) ----------
__global__ __launch_bounds__(256) void uhat_kernel(const float* __restrict__ Wr,
                                                   const float* __restrict__ u,
                                                   float* __restrict__ u_hat){
  __shared__ float su[4][64];                 // [lp-pair in block][b*16+q]
  int lp0 = blockIdx.x * 4;
  {
    int t = threadIdx.x, pr = t >> 6, rest = t & 63, bb = rest >> 4, q = rest & 15;
    int pp = (lp0 + pr) & (P_ - 1);
    su[pr][rest] = u[((size_t)bb*P_ + pp)*Q_ + q];
  }
  __syncthreads();
  size_t flat = (size_t)blockIdx.x*256 + threadIdx.x;   // (l*P+p)*V + v
  int pair = threadIdx.x >> 6;
  const float4* w4 = (const float4*)(Wr + flat*Q_);
  float4 wa = w4[0], wb = w4[1], wc = w4[2], wd = w4[3];
  #pragma unroll
  for (int bb = 0; bb < B_; ++bb){
    const float* s = &su[pair][bb*16];
    float acc = wa.x*s[0] + wa.y*s[1] + wa.z*s[2]  + wa.w*s[3]
              + wb.x*s[4] + wb.y*s[5] + wb.z*s[6]  + wb.w*s[7]
              + wc.x*s[8] + wc.y*s[9] + wc.z*s[10] + wc.w*s[11]
              + wd.x*s[12]+ wd.y*s[13]+ wd.z*s[14] + wd.w*s[15];
    u_hat[(size_t)bb*(L_*P_*V_) + flat] = acc;
  }
}

// ---------- routing: softmax over L ----------
__global__ void route_softmax_kernel(const float* __restrict__ b_ij, float* __restrict__ c_ij){
  int i = blockIdx.x*blockDim.x + threadIdx.x;
  if (i >= B_*P_) return;
  int b = i / P_, p = i % P_;
  size_t base = (size_t)b*L_*P_ + p;
  float mx = -1e30f;
  for (int l = 0; l < L_; ++l) mx = fmaxf(mx, b_ij[base + (size_t)l*P_]);
  float s = 0.f;
  for (int l = 0; l < L_; ++l) s += expf(b_ij[base + (size_t)l*P_] - mx);
  float inv = 1.f / s;
  for (int l = 0; l < L_; ++l) c_ij[base + (size_t)l*P_] = expf(b_ij[base + (size_t)l*P_] - mx) * inv;
}

// ---------- routing: s = sum_p c*u_hat, squash -> v_j (one block per (b,l)) ----------
__global__ __launch_bounds__(256) void route_agg_kernel(const float* __restrict__ c_ij,
                                                        const float* __restrict__ u_hat,
                                                        float* __restrict__ v_j){
  int bl = blockIdx.x;
  int v = threadIdx.x & 63, chunk = threadIdx.x >> 6;
  size_t ub = (size_t)bl*P_*V_;
  size_t cb = (size_t)bl*P_;
  float acc = 0.f;
  int p0 = chunk*256;
  for (int p = p0; p < p0 + 256; ++p)
    acc = fmaf(c_ij[cb + p], u_hat[ub + (size_t)p*V_ + v], acc);
  __shared__ float red[256];
  __shared__ float sscale;
  red[threadIdx.x] = acc; __syncthreads();
  if (threadIdx.x < 64) red[v] = red[v] + red[64+v] + red[128+v] + red[192+v];
  __syncthreads();
  if (threadIdx.x == 0){
    float sn = 0.f;
    for (int k = 0; k < 64; ++k) sn += red[k]*red[k];
    sscale = sn / ((1.f + sn) * sqrtf(sn));
  }
  __syncthreads();
  if (threadIdx.x < 64) v_j[(size_t)bl*V_ + v] = sscale * red[v];
}

// ---------- routing: b_ij += sum_v v_j*u_hat ----------
__global__ __launch_bounds__(256) void route_update_kernel(const float* __restrict__ v_j,
                                                           const float* __restrict__ u_hat,
                                                           float* __restrict__ b_ij){
  size_t idx = (size_t)blockIdx.x*256 + threadIdx.x;   // (b*L+l)*P + p
  int bl = (int)(idx / P_);                            // whole block shares bl (P%256==0)
  __shared__ float vj[64];
  if (threadIdx.x < 64) vj[threadIdx.x] = v_j[(size_t)bl*V_ + threadIdx.x];
  __syncthreads();
  const float4* uh = (const float4*)(u_hat + idx*V_);
  float acc = 0.f;
  #pragma unroll
  for (int k = 0; k < 16; ++k){
    float4 t = uh[k];
    acc += t.x*vj[4*k] + t.y*vj[4*k+1] + t.z*vj[4*k+2] + t.w*vj[4*k+3];
  }
  b_ij[idx] += acc;
}

// ---------- head: logits + presence ----------
__global__ void head_kernel(const float* __restrict__ v_j, const float* __restrict__ fc_w,
                            const float* __restrict__ fc_b, float* __restrict__ out){
  int id = blockIdx.x*blockDim.x + threadIdx.x;
  if (id < B_*OUT_){
    int b = id / OUT_, o = id % OUT_;
    const float* vb = v_j + (size_t)b*L_*V_;
    const float* w  = fc_w + (size_t)o*L_*V_;
    float acc = fc_b[o];
    for (int k = 0; k < L_*V_; ++k) acc = fmaf(vb[k], w[k], acc);
    out[b*OUT_ + o] = acc;
  } else if (id < B_*OUT_ + B_*L_){
    int j = id - B_*OUT_;
    const float* vr = v_j + (size_t)j*V_;
    float sn = 0.f;
    for (int k = 0; k < V_; ++k) sn += vr[k]*vr[k];
    out[B_*OUT_ + j] = sqrtf(sn);
  }
}

// ---------- host ----------
extern "C" void kernel_launch(void* const* d_in, const int* in_sizes, int n_in,
                              void* d_out, int out_size, void* d_ws, size_t ws_size,
                              hipStream_t stream) {
  (void)in_sizes; (void)n_in; (void)out_size; (void)ws_size;
  const float* x    = (const float*)d_in[0];
  const float* w1   = (const float*)d_in[1];
  const float* b1   = (const float*)d_in[2];
  const float* g1   = (const float*)d_in[3];
  const float* be1  = (const float*)d_in[4];
  const float* w2   = (const float*)d_in[5];
  const float* b2   = (const float*)d_in[6];
  const float* g2   = (const float*)d_in[7];
  const float* be2  = (const float*)d_in[8];
  const float* w3   = (const float*)d_in[9];
  /* b3 (d_in[10]) cancels under train-mode BN */
  const float* g3   = (const float*)d_in[11];
  const float* be3  = (const float*)d_in[12];
  const float* Wr   = (const float*)d_in[13];
  const float* fc_w = (const float*)d_in[14];
  const float* fc_b = (const float*)d_in[15];

  char* wsp = (char*)d_ws;
  auto alloc = [&](size_t bytes) -> void* {
    void* p = (void*)wsp; wsp += (bytes + 255) & ~(size_t)255; return p;
  };
  float*          h1pre = (float*)alloc((size_t)B_*64*N_*4);
  float*          h1    = (float*)alloc((size_t)B_*64*N_*4);
  float*          h2pre = (float*)alloc((size_t)B_*128*N_*4);
  unsigned short* hTbf  = (unsigned short*)alloc((size_t)B_*N_*128*2);
  unsigned short* w3bf  = (unsigned short*)alloc((size_t)M_*128*2);
  float* sum1 = (float*)alloc(64*4);  float* sq1 = (float*)alloc(64*4);
  float* a1   = (float*)alloc(64*4);  float* c1  = (float*)alloc(64*4);
  float* sum2 = (float*)alloc(128*4); float* sq2 = (float*)alloc(128*4);
  float* a2   = (float*)alloc(128*4); float* c2  = (float*)alloc(128*4);
  float* sum3 = (float*)alloc((size_t)M_*4);     float* sq3 = (float*)alloc((size_t)M_*4);
  float* mx3  = (float*)alloc((size_t)B_*M_*4);  float* mn3 = (float*)alloc((size_t)B_*M_*4);
  float* u    = (float*)alloc((size_t)B_*P_*Q_*4);
  float* u_hat= (float*)alloc((size_t)B_*L_*P_*V_*4);
  float* b_ij = (float*)alloc((size_t)B_*L_*P_*4);
  float* c_ij = (float*)alloc((size_t)B_*L_*P_*4);
  float* v_j  = (float*)alloc((size_t)B_*L_*V_*4);

  // zero-init accumulators and routing logits
  hipMemsetAsync(sum1, 0, 64*4, stream);  hipMemsetAsync(sq1, 0, 64*4, stream);
  hipMemsetAsync(sum2, 0, 128*4, stream); hipMemsetAsync(sq2, 0, 128*4, stream);
  hipMemsetAsync(sum3, 0, (size_t)M_*4, stream);
  hipMemsetAsync(sq3,  0, (size_t)M_*4, stream);
  hipMemsetAsync(b_ij, 0, (size_t)B_*L_*P_*4, stream);
  fill_u32_kernel<<<(B_*M_+255)/256, 256, 0, stream>>>((unsigned*)mx3, 0xFF800000u, B_*M_); // -inf
  fill_u32_kernel<<<(B_*M_+255)/256, 256, 0, stream>>>((unsigned*)mn3, 0x7F800000u, B_*M_); // +inf

  // stage 1
  conv1_kernel<<<dim3(N_/256, 64, B_), 256, 0, stream>>>(x, w1, b1, h1pre, sum1, sq1);
  bn_finalize_kernel<<<1, 64, 0, stream>>>(sum1, sq1, g1, be1, a1, c1, 64);
  apply1_kernel<<<(B_*64*N_)/256, 256, 0, stream>>>(h1pre, a1, c1, h1);
  conv2_kernel<<<dim3(N_/256, 128, B_), 256, 0, stream>>>(h1, w2, b2, h2pre, sum2, sq2);
  bn_finalize_kernel<<<1, 128, 0, stream>>>(sum2, sq2, g2, be2, a2, c2, 128);
  apply2t_kernel<<<B_*N_, 128, 0, stream>>>(h2pre, a2, c2, hTbf);
  f32_to_bf16_kernel<<<(M_*128)/256, 256, 0, stream>>>(w3, w3bf, M_*128);

  // stage 2: WMMA GEMM with fused BN-stat + maxpool epilogue (u never stored)
  caps_gemm_kernel<<<4096, 256, 0, stream>>>(w3bf, hTbf, sum3, sq3, mx3, mn3);
  caps_squash_kernel<<<(B_*P_+255)/256, 256, 0, stream>>>(sum3, sq3, g3, be3, mx3, mn3, u);

  // stage 3: u_hat + dynamic routing (3 iters)
  uhat_kernel<<<(L_*P_*V_)/256, 256, 0, stream>>>(Wr, u, u_hat);
  for (int it = 0; it < 3; ++it){
    route_softmax_kernel<<<(B_*P_+255)/256, 256, 0, stream>>>(b_ij, c_ij);
    route_agg_kernel<<<B_*L_, 256, 0, stream>>>(c_ij, u_hat, v_j);
    if (it < 2)
      route_update_kernel<<<(B_*L_*P_)/256, 256, 0, stream>>>(v_j, u_hat, b_ij);
  }

  // head: logits [4,40] then presence [4,64]
  head_kernel<<<2, 256, 0, stream>>>(v_j, fc_w, fc_b, (float*)d_out);
}
